// GRU_62947040690689
// MI455X (gfx1250) — matrix operand order repaired and checked by reading, not compiled, not run on hardware
//
#include <hip/hip_runtime.h>
#include <hip/hip_bf16.h>

typedef __attribute__((ext_vector_type(16))) __bf16 v16bf;
typedef __attribute__((ext_vector_type(8)))  float  v8f;

union FragAB { v16bf v; uint4 q[2]; unsigned short s[16]; };

constexpr int Bsz = 4096;   // batch
constexpr int Isz = 64;     // input/output features
constexpr int Hsz = 256;    // hidden
constexpr int Tsz = 64;     // n_frames (fixed by setup_inputs)

__device__ __forceinline__ unsigned short f2bf(float x) {
  unsigned int u = __float_as_uint(x);
  return (unsigned short)((u + 0x7FFFu + ((u >> 16) & 1u)) >> 16);  // RNE
}
__device__ __forceinline__ v8f splat8(float x) {
  v8f v;
#pragma unroll
  for (int i = 0; i < 8; ++i) v[i] = x;
  return v;
}
__device__ __forceinline__ float sigm(float x) { return 1.f / (1.f + __expf(-x)); }
__device__ __forceinline__ float tanh_fast(float x) {
  float e = __expf(-2.f * x);
  return (1.f - e) / (1.f + e);
}

// A fragment: 16x32 bf16 tile from row-major [16][ld] ushort buffer (LDS).
// ISA layout: lanes 0-15 hold M=lane, K=(0..7,16..23); lanes 16-31 hold K=(8..15,24..31).
__device__ __forceinline__ v16bf load_A(const unsigned short* p, int ld) {
  const int lane = threadIdx.x & 31;
  const int m = lane & 15, hi = lane >> 4;
  FragAB f;
  f.q[0] = *(const uint4*)(p + m * ld + hi * 8);
  f.q[1] = *(const uint4*)(p + m * ld + hi * 8 + 16);
  return f.v;
}
// B fragment: 32x16 bf16 tile; weights stored row-major [N][ld] (row = output col n,
// cols = k). Lanes 0-15 hold K=(0..7,16..23), lanes 16-31 hold K=(8..15,24..31), N = lane&15.
__device__ __forceinline__ v16bf load_B(const unsigned short* p, int ld) {
  const int lane = threadIdx.x & 31;
  const int n = lane & 15, hi = lane >> 4;
  const unsigned short* q = p + n * ld + hi * 8;
  FragAB f;
  f.q[0] = *(const uint4*)(q);
  f.q[1] = *(const uint4*)(q + 16);
  return f.v;
}
__device__ __forceinline__ v8f wmma_bf16(v16bf a, v16bf b, v8f c) {
  // (neg_a, A, neg_b, B, c_mod, C, reuse_a, reuse_b)
  return __builtin_amdgcn_wmma_f32_16x16x32_bf16(false, a, false, b, (short)0, c,
                                                 false, false);
}

// ---------------------------------------------------------------------------
// Kernel 1: spectral-norm sigma (one power iteration, torch semantics) +
// bf16 conversion of the three weight matrices into workspace.
// ---------------------------------------------------------------------------
__global__ __launch_bounds__(256) void sn_prep_kernel(
    const float* __restrict__ w_ih, const float* __restrict__ w_hh,
    const float* __restrict__ w_lin, const float* __restrict__ u_sn,
    unsigned short* __restrict__ w_ih_bf, unsigned short* __restrict__ w_hh_bf,
    unsigned short* __restrict__ w_sn_bf) {
  __shared__ float red[256];
  __shared__ float vsh[Hsz];
  __shared__ float sig;
  const int tid = threadIdx.x;

  // v_j = (W^T u)_j, j in [0,256)
  float vj = 0.f;
  for (int i = 0; i < Isz; ++i) vj += w_lin[i * Hsz + tid] * u_sn[i];
  red[tid] = vj * vj;
  __syncthreads();
  for (int s = 128; s > 0; s >>= 1) {
    if (tid < s) red[tid] += red[tid + s];
    __syncthreads();
  }
  float vnorm = sqrtf(red[0]) + 1e-12f;
  __syncthreads();
  vsh[tid] = vj / vnorm;
  __syncthreads();

  // wv_i = (W v)_i, i in [0,64);  sigma = u.(Wv) with u = Wv/(||Wv||+eps)
  float wv = 0.f;
  if (tid < Isz)
    for (int j = 0; j < Hsz; ++j) wv += w_lin[tid * Hsz + j] * vsh[j];
  red[tid] = (tid < Isz) ? wv * wv : 0.f;
  __syncthreads();
  for (int s = 128; s > 0; s >>= 1) {
    if (tid < s) red[tid] += red[tid + s];
    __syncthreads();
  }
  if (tid == 0) {
    float n2 = red[0];
    sig = n2 / (sqrtf(n2) + 1e-12f);
  }
  __syncthreads();
  const float inv_sigma = 1.f / sig;

  for (int idx = tid; idx < 3 * Hsz * Isz; idx += 256) w_ih_bf[idx] = f2bf(w_ih[idx]);
  for (int idx = tid; idx < 3 * Hsz * Hsz; idx += 256) w_hh_bf[idx] = f2bf(w_hh[idx]);
  for (int idx = tid; idx < Isz * Hsz; idx += 256)
    w_sn_bf[idx] = f2bf(w_lin[idx] * inv_sigma);
}

// ---------------------------------------------------------------------------
// Kernel 2: the recurrence. One workgroup (4 waves) owns a 16-row batch tile.
// Waves split the 256 hidden columns 4-ways; h and x are double-buffered bf16
// in LDS; the wave's own h columns additionally persist in registers (C
// layout) so the z*h blend is exact fp32. A fragments are re-read from LDS
// inside the k loops (cheap ds_load_b128) instead of pinning 80 VGPRs, and
// k-loop unrolling is capped so the scheduler cannot hoist every B fragment.
// ---------------------------------------------------------------------------
__global__ __launch_bounds__(128) void gru_wmma_kernel(
    const float* __restrict__ inputs,
    const unsigned short* __restrict__ w_ih_bf,
    const unsigned short* __restrict__ w_hh_bf,
    const unsigned short* __restrict__ w_sn_bf,
    const float* __restrict__ b_ih, const float* __restrict__ b_hh,
    const float* __restrict__ b_lin, float* __restrict__ out) {
  __shared__ unsigned short h_buf[2][16][Hsz];  // 16 KB
  __shared__ unsigned short x_buf[2][16][Isz];  //  4 KB
  const int wave = threadIdx.x >> 5;  // 0..3
  const int lane = threadIdx.x & 31;
  const int n16 = lane & 15, hi = lane >> 4;
  const int b0 = blockIdx.x * 16;

  // h0 = 0; x0 = inputs (bf16)
  for (int idx = threadIdx.x; idx < 16 * Hsz; idx += 128)
    h_buf[0][idx >> 8][idx & (Hsz - 1)] = 0;
  for (int idx = threadIdx.x; idx < 16 * Isz; idx += 128)
    x_buf[0][idx >> 6][idx & (Isz - 1)] =
        f2bf(inputs[(size_t)(b0 + (idx >> 6)) * Isz + (idx & (Isz - 1))]);
  __syncthreads();

  // Per-wave fp32 copy of its own hidden columns, C layout: hreg[tt][r] is
  // h[m = r + hi*8][ (wave*4+tt)*16 + n16 ].
  v8f hreg[4];
#pragma unroll
  for (int tt = 0; tt < 4; ++tt) hreg[tt] = splat8(0.f);

  int p = 0;
  for (int t = 0; t < Tsz; ++t) {
    const int pp = p ^ 1;

    for (int tt = 0; tt < 4; ++tt) {
      const int n0 = (wave * 4 + tt) * 16;
      const int c = n0 + n16;
      v8f accr  = splat8(b_ih[c] + b_hh[c]);
      v8f accz  = splat8(b_ih[c + Hsz] + b_hh[c + Hsz]);
      v8f accin = splat8(b_ih[c + 2 * Hsz]);
      v8f acchn = splat8(b_hh[c + 2 * Hsz]);

#pragma unroll 2
      for (int k = 0; k < 2; ++k) {  // gi contributions, K = 64
        v16bf xa = load_A(&x_buf[p][0][0] + k * 32, Isz);
        accr  = wmma_bf16(xa, load_B(w_ih_bf + (size_t)n0 * Isz + k * 32, Isz), accr);
        accz  = wmma_bf16(xa, load_B(w_ih_bf + (size_t)(n0 + Hsz) * Isz + k * 32, Isz), accz);
        accin = wmma_bf16(xa, load_B(w_ih_bf + (size_t)(n0 + 2 * Hsz) * Isz + k * 32, Isz), accin);
      }
#pragma unroll 2
      for (int k = 0; k < 8; ++k) {  // gh contributions, K = 256
        v16bf ha = load_A(&h_buf[p][0][0] + k * 32, Hsz);
        accr  = wmma_bf16(ha, load_B(w_hh_bf + (size_t)n0 * Hsz + k * 32, Hsz), accr);
        accz  = wmma_bf16(ha, load_B(w_hh_bf + (size_t)(n0 + Hsz) * Hsz + k * 32, Hsz), accz);
        acchn = wmma_bf16(ha, load_B(w_hh_bf + (size_t)(n0 + 2 * Hsz) * Hsz + k * 32, Hsz), acchn);
      }

      // Gate math on the C layout: element r -> (m = r + hi*8, n = n16).
      v8f hnew;
#pragma unroll
      for (int r = 0; r < 8; ++r) {
        float rg = sigm(accr[r]);
        float zg = sigm(accz[r]);
        float ng = tanh_fast(accin[r] + rg * acchn[r]);
        hnew[r] = (1.f - zg) * ng + zg * hreg[tt][r];
      }
      hreg[tt] = hnew;
#pragma unroll
      for (int r = 0; r < 8; ++r)
        h_buf[pp][r + hi * 8][n0 + n16] = f2bf(hnew[r]);
    }
    __syncthreads();  // full h_new visible

    // y = h_new @ w_sn^T + b_lin; wave w produces output cols [16w, 16w+16).
    {
      const int n0 = wave * 16;
      v8f acc = splat8(b_lin[n0 + n16]);
#pragma unroll 2
      for (int k = 0; k < 8; ++k) {
        v16bf hna = load_A(&h_buf[pp][0][0] + k * 32, Hsz);
        acc = wmma_bf16(hna, load_B(w_sn_bf + (size_t)n0 * Hsz + k * 32, Hsz), acc);
      }
#pragma unroll
      for (int r = 0; r < 8; ++r) {
        const int m = r + hi * 8;
        out[((size_t)t * Bsz + b0 + m) * Isz + n0 + n16] = acc[r];  // raw y (pre-BN)
        x_buf[pp][m][n0 + n16] = f2bf(acc[r]);                      // feedback input
      }
    }
    __syncthreads();  // x_buf[pp] complete
    p = pp;
  }
}

// ---------------------------------------------------------------------------
// Kernel 3: BatchNorm1d(affine=False), training stats, per timestep, in place.
// Safe in place: gru_wmma_kernel fully rewrites d_out on every launch.
// ---------------------------------------------------------------------------
__global__ __launch_bounds__(256) void bn_kernel(float* __restrict__ out) {
  __shared__ float ssum[256], ssq[256];
  __shared__ float smean[Isz], sinv[Isz];
  const int t = blockIdx.x;
  const int o = threadIdx.x & 63;
  const int g = threadIdx.x >> 6;  // 4 batch groups
  const size_t base = (size_t)t * Bsz * Isz;

  float s = 0.f, q = 0.f;
  for (int b = g; b < Bsz; b += 4) {
    float v = out[base + (size_t)b * Isz + o];
    s += v;
    q += v * v;
  }
  ssum[threadIdx.x] = s;
  ssq[threadIdx.x] = q;
  __syncthreads();
  if (g == 0) {
    for (int gg = 1; gg < 4; ++gg) {
      s += ssum[gg * 64 + o];
      q += ssq[gg * 64 + o];
    }
    float mean = s * (1.f / Bsz);
    float var = q * (1.f / Bsz) - mean * mean;  // biased variance
    smean[o] = mean;
    sinv[o] = 1.f / sqrtf(var + 1e-5f);
  }
  __syncthreads();
  for (int idx = threadIdx.x; idx < Bsz * Isz; idx += 256) {
    size_t off = base + idx;
    out[off] = (out[off] - smean[idx & 63]) * sinv[idx & 63];
  }
}

// ---------------------------------------------------------------------------
extern "C" void kernel_launch(void* const* d_in, const int* in_sizes, int n_in,
                              void* d_out, int out_size, void* d_ws, size_t ws_size,
                              hipStream_t stream) {
  (void)in_sizes; (void)n_in; (void)out_size; (void)ws_size;
  const float* inputs = (const float*)d_in[0];
  const float* w_ih   = (const float*)d_in[1];
  const float* w_hh   = (const float*)d_in[2];
  const float* b_ih   = (const float*)d_in[3];
  const float* b_hh   = (const float*)d_in[4];
  const float* w_lin  = (const float*)d_in[5];
  const float* b_lin  = (const float*)d_in[6];
  const float* u_sn   = (const float*)d_in[7];
  // d_in[8] = n_frames (device scalar); grid config must be static -> T=64.
  float* out = (float*)d_out;

  // Workspace: bf16 weights (512 KB total).
  unsigned short* w_ih_bf = (unsigned short*)d_ws;        // 3*256*64  = 49152
  unsigned short* w_hh_bf = w_ih_bf + 3 * Hsz * Isz;      // 3*256*256 = 196608
  unsigned short* w_sn_bf = w_hh_bf + 3 * Hsz * Hsz;      // 64*256    = 16384

  sn_prep_kernel<<<1, 256, 0, stream>>>(w_ih, w_hh, w_lin, u_sn,
                                        w_ih_bf, w_hh_bf, w_sn_bf);
  gru_wmma_kernel<<<Bsz / 16, 128, 0, stream>>>(inputs, w_ih_bf, w_hh_bf, w_sn_bf,
                                                b_ih, b_hh, b_lin, out);
  bn_kernel<<<Tsz, 256, 0, stream>>>(out);
}
